// AttentionHead_45028437131996
// MI455X (gfx1250) — matrix operand order repaired and checked by reading, not compiled, not run on hardware
//
#include <hip/hip_runtime.h>
#include <hip/hip_bf16.h>

typedef __attribute__((ext_vector_type(16))) __bf16 bh16;
typedef __attribute__((ext_vector_type(8)))  __bf16 bh8;
typedef __attribute__((ext_vector_type(8)))  float  f32x8;

#define BATCH 4
#define T_SEQ 4096
#define CDIM  1024
#define HDIM  64

__device__ inline bh16 cat8(bh8 lo, bh8 hi) {
  return __builtin_shufflevector(lo, hi, 0,1,2,3,4,5,6,7,8,9,10,11,12,13,14,15);
}
__device__ inline bh8 ldg8(const __bf16* p) { return *reinterpret_cast<const bh8*>(p); }

__device__ inline float redmax16(float v) {
  #pragma unroll
  for (int m = 8; m >= 1; m >>= 1) v = fmaxf(v, __shfl_xor(v, m, 32));
  return v;
}
__device__ inline float redsum16(float v) {
  #pragma unroll
  for (int m = 8; m >= 1; m >>= 1) v += __shfl_xor(v, m, 32);
  return v;
}

// CDNA5 async global->LDS copy (ASYNCcnt-tracked; 16B per lane).
// Generic pointers to __shared__ carry the LDS byte offset in their low 32 bits.
__device__ inline void async_copy_b128(__bf16* ldst, const __bf16* gsrc) {
  unsigned lds_off = (unsigned)(uintptr_t)ldst;
  asm volatile("global_load_async_to_lds_b128 %0, %1, off"
               :: "v"(lds_off), "v"(gsrc) : "memory");
}
// Wait until at most N async ops outstanding (in-order completion => the
// oldest ones have landed).
__device__ inline void wait_async0() {
  asm volatile("s_wait_asynccnt 0x0" ::: "memory");
}
__device__ inline void wait_async4() {
  asm volatile("s_wait_asynccnt 0x4" ::: "memory");
}

// ---------------- Kernel 0: transpose + bf16-convert weights --------------
__global__ __launch_bounds__(256) void wtrans(const float* __restrict__ Wk,
                                              const float* __restrict__ Wq,
                                              const float* __restrict__ Wv,
                                              __bf16* __restrict__ Wkt,
                                              __bf16* __restrict__ Wqt,
                                              __bf16* __restrict__ Wvt) {
  int gid = blockIdx.x * 256 + threadIdx.x;   // 3 * 65536 total
  int w = gid >> 16;
  int r = gid & 65535;
  int c = r >> 6, h = r & 63;
  const float* src = (w == 0) ? Wk : (w == 1) ? Wq : Wv;
  __bf16* dst = (w == 0) ? Wkt : (w == 1) ? Wqt : Wvt;
  float scale = (w == 1) ? 0.03125f : 1.0f;   // 1024^-0.5 folded into Wq
  dst[(size_t)h * CDIM + c] = (__bf16)(src[(size_t)c * HDIM + h] * scale);
}

// ---------------- Kernel 1: QKV projection via bf16 WMMA -------------------
// One wave per 16-row tile of x; single pass, 12 accumulators. B-tile loads
// are batched ahead of the WMMA group so loads overlap matrix issue.
__global__ __launch_bounds__(128) void qkv_proj(const float* __restrict__ x,
    const __bf16* __restrict__ Wkt, const __bf16* __restrict__ Wqt,
    const __bf16* __restrict__ Wvt,
    __bf16* __restrict__ Kb, __bf16* __restrict__ Qb, __bf16* __restrict__ Vt)
{
  const int lane = threadIdx.x & 31;
  const int wid  = threadIdx.x >> 5;
  const int wave = blockIdx.x * 4 + wid;       // 1024 waves total
  const int b    = wave >> 8;                  // 256 tiles per batch
  const int t0   = (wave & 255) << 4;
  const int l15  = lane & 15;
  const bool hiL = lane >= 16;
  const int off1 = hiL ? 8 : 0;                // ISA 16-bit A-layout K split
  const int rowb = hiL ? 8 : 0;                // ISA C/D layout row split

  const float* xrow = x + ((size_t)(b * T_SEQ + t0 + l15)) * CDIM;
  const __bf16* const Ws[3] = { Wkt, Wqt, Wvt };

  f32x8 acc[3][4] = {};
  for (int c0 = 0; c0 < CDIM; c0 += 32) {
    // A tile: lane<16 holds K {c0+0..7, c0+16..23}; lane>=16 {+8, +24}
    const float4 f0 = *reinterpret_cast<const float4*>(xrow + c0 + off1);
    const float4 f1 = *reinterpret_cast<const float4*>(xrow + c0 + off1 + 4);
    const float4 f2 = *reinterpret_cast<const float4*>(xrow + c0 + off1 + 16);
    const float4 f3 = *reinterpret_cast<const float4*>(xrow + c0 + off1 + 20);
    bh16 a;
    a[0]=(__bf16)f0.x;  a[1]=(__bf16)f0.y;  a[2]=(__bf16)f0.z;  a[3]=(__bf16)f0.w;
    a[4]=(__bf16)f1.x;  a[5]=(__bf16)f1.y;  a[6]=(__bf16)f1.z;  a[7]=(__bf16)f1.w;
    a[8]=(__bf16)f2.x;  a[9]=(__bf16)f2.y;  a[10]=(__bf16)f2.z; a[11]=(__bf16)f2.w;
    a[12]=(__bf16)f3.x; a[13]=(__bf16)f3.y; a[14]=(__bf16)f3.z; a[15]=(__bf16)f3.w;

    #pragma unroll
    for (int w = 0; w < 3; ++w) {
      const __bf16* wb = Ws[w] + (size_t)c0 + (hiL ? 8 : 0);
      bh16 bm[4];
      #pragma unroll
      for (int j = 0; j < 4; ++j) {              // batch loads first
        const __bf16* p = wb + (size_t)(16 * j + l15) * CDIM;
        bm[j] = cat8(ldg8(p), ldg8(p + 16));
      }
      #pragma unroll
      for (int j = 0; j < 4; ++j)                // then the WMMA group
        acc[w][j] = __builtin_amdgcn_wmma_f32_16x16x32_bf16(
            false, a, false, bm[j], (short)0, acc[w][j], false, false);
    }
  }

  // Store from C/D layout (VGPR r: lanes<16 -> row r, lanes>=16 -> row r+8)
  #pragma unroll
  for (int j = 0; j < 4; ++j) {
    #pragma unroll
    for (int r = 0; r < 8; ++r) {
      const int gr = t0 + rowb + r;
      const int gc = 16 * j + l15;
      Kb[((size_t)(b * T_SEQ + gr)) * HDIM + gc] = (__bf16)acc[0][j][r];
      Qb[((size_t)(b * T_SEQ + gr)) * HDIM + gc] = (__bf16)acc[1][j][r];
      Vt[((size_t)(b * HDIM + gc)) * T_SEQ + gr] = (__bf16)acc[2][j][r]; // V^T
    }
  }
}

// ---------------- Kernel 2: fused causal flash attention -------------------
// 4 waves/WG cover 64 Q rows. K/V blocks are double-buffered in LDS via
// async copies so block n+1 streams from HBM while block n is computed.
__global__ __launch_bounds__(128) void attn(const __bf16* __restrict__ Qb,
                                            const __bf16* __restrict__ Kb,
                                            const __bf16* __restrict__ Vt,
                                            float* __restrict__ out)
{
  __shared__ __bf16 Klds[2][32 * 64];          // 2 x 4KB: K block [s][h]
  __shared__ __bf16 Vlds[2][64 * 32];          // 2 x 4KB: V block [h][s]
  __shared__ __bf16 Plds[4][16 * 32];          // per-wave P transpose slab

  const int tid  = threadIdx.x;
  const int lane = tid & 31;
  const int wid  = tid >> 5;
  const int b    = blockIdx.x >> 6;            // 64 WGs per batch
  const int t0wg = (blockIdx.x & 63) << 6;     // 64 rows per WG
  const int t0   = t0wg + (wid << 4);          // 16 rows per wave
  const int l15  = lane & 15;
  const bool hiL = lane >= 16;
  const int off1 = hiL ? 8 : 0;
  const int rowb = hiL ? 8 : 0;
  const float NEG_INF = -__builtin_inff();

  // Persistent Q A-tiles (16x64 bf16 across two 16x32 tiles)
  const __bf16* qrow = Qb + ((size_t)(b * T_SEQ + t0 + l15)) * HDIM;
  bh16 qa[2];
  #pragma unroll
  for (int hc = 0; hc < 2; ++hc)
    qa[hc] = cat8(ldg8(qrow + 32 * hc + off1), ldg8(qrow + 32 * hc + off1 + 16));

  f32x8 O[4] = {};
  float mrow[8], lrow[8];
  #pragma unroll
  for (int r = 0; r < 8; ++r) { mrow[r] = NEG_INF; lrow[r] = 0.0f; }

  const __bf16* kbase = Kb + (size_t)b * T_SEQ * HDIM;
  const __bf16* vbase = Vt + (size_t)b * HDIM * T_SEQ;
  __bf16* pl = &Plds[wid][0];

  // Per-thread staging assignments (128 threads x 16B x 2 issues per buffer)
  const int kcopy = tid * 8;                   // bf16 elems within 4KB K block
  const int vh    = tid >> 2;                  // 0..31 (h rows; +32 second half)
  const int vpart = (tid & 3) * 8;             // elems within a 32-elem V row

  auto stage = [&](int buf, int s0) {
    async_copy_b128(&Klds[buf][kcopy],        kbase + (size_t)s0 * HDIM + kcopy);
    async_copy_b128(&Klds[buf][kcopy + 1024], kbase + (size_t)s0 * HDIM + kcopy + 1024);
    async_copy_b128(&Vlds[buf][vh * 32 + vpart],
                    vbase + (size_t)vh * T_SEQ + s0 + vpart);
    async_copy_b128(&Vlds[buf][(vh + 32) * 32 + vpart],
                    vbase + (size_t)(vh + 32) * T_SEQ + s0 + vpart);
  };

  const int nblk = (t0wg + 95) >> 5;           // ceil((t0wg+64)/32)
  stage(0, 0);                                 // prologue: first block
  for (int blk = 0; blk < nblk; ++blk) {
    const int s0  = blk << 5;
    const int cur = blk & 1;

    if (blk + 1 < nblk) {                      // WG-uniform branch
      stage(cur ^ 1, s0 + 32);                 // overlap next block's copies
      __builtin_prefetch(kbase + (size_t)(s0 + 64) * HDIM + kcopy, 0, 0);
      wait_async4();                           // newest 4 may stay in flight
    } else {
      wait_async0();
    }
    __syncthreads();

    // S = Q * K^T  (two 16x16 tiles; B columns = sequence index)
    const __bf16* Kl = &Klds[cur][0];
    const __bf16* Vl = &Vlds[cur][0];
    f32x8 S[2] = {};
    #pragma unroll
    for (int tile = 0; tile < 2; ++tile) {
      const __bf16* kp = Kl + (16 * tile + l15) * HDIM + (hiL ? 8 : 0);
      bh16 kb0 = cat8(ldg8(kp),      ldg8(kp + 16));  // h 0..31 chunk
      bh16 kb1 = cat8(ldg8(kp + 32), ldg8(kp + 48));  // h 32..63 chunk
      S[tile] = __builtin_amdgcn_wmma_f32_16x16x32_bf16(
          false, qa[0], false, kb0, (short)0, S[tile], false, false);
      S[tile] = __builtin_amdgcn_wmma_f32_16x16x32_bf16(
          false, qa[1], false, kb1, (short)0, S[tile], false, false);
    }

    // Causal mask + online softmax (fully-masked blocks: p=0, alpha=1)
    #pragma unroll
    for (int r = 0; r < 8; ++r) {
      const int grow = t0 + rowb + r;
      const int c0i  = s0 + l15;
      float s0v = (c0i      <= grow) ? S[0][r] : NEG_INF;
      float s1v = (c0i + 16 <= grow) ? S[1][r] : NEG_INF;
      float t   = redmax16(fmaxf(s0v, s1v));
      float mn  = fmaxf(mrow[r], t);
      float alpha = __expf(mrow[r] - mn);
      mrow[r] = mn;
      float p0 = __expf(s0v - mn);
      float p1 = __expf(s1v - mn);
      lrow[r] = lrow[r] * alpha + redsum16(p0 + p1);
      #pragma unroll
      for (int j = 0; j < 4; ++j) O[j][r] *= alpha;
      pl[(rowb + r) * 32 + l15]      = (__bf16)p0;   // C/D -> row-major LDS
      pl[(rowb + r) * 32 + l15 + 16] = (__bf16)p1;
    }

    // Re-read P in A-matrix layout from LDS
    const __bf16* pr = pl + l15 * 32;
    bh16 pa = cat8(*reinterpret_cast<const bh8*>(pr + off1),
                   *reinterpret_cast<const bh8*>(pr + off1 + 16));

    // O += P * V   (B from staged V^T: contiguous along sequence)
    #pragma unroll
    for (int j = 0; j < 4; ++j) {
      const __bf16* vp = Vl + (16 * j + l15) * 32 + (hiL ? 8 : 0);
      bh16 vb = cat8(ldg8(vp), ldg8(vp + 16));
      O[j] = __builtin_amdgcn_wmma_f32_16x16x32_bf16(
          false, pa, false, vb, (short)0, O[j], false, false);
    }
    __syncthreads();   // nobody overwrites 'cur' until all waves finished it
  }

  // Normalize and store f32 output (coalesced along h)
  float* orow = out + ((size_t)(b * T_SEQ + t0 + rowb)) * HDIM;
  #pragma unroll
  for (int r = 0; r < 8; ++r) {
    const float inv = 1.0f / lrow[r];
    #pragma unroll
    for (int j = 0; j < 4; ++j)
      orow[(size_t)r * HDIM + 16 * j + l15] = O[j][r] * inv;
  }
}

extern "C" void kernel_launch(void* const* d_in, const int* in_sizes, int n_in,
                              void* d_out, int out_size, void* d_ws, size_t ws_size,
                              hipStream_t stream) {
  const float* x  = (const float*)d_in[0];
  const float* Wk = (const float*)d_in[1];
  const float* Wq = (const float*)d_in[2];
  const float* Wv = (const float*)d_in[3];
  float* out = (float*)d_out;

  // Workspace layout (bf16): Wkt|Wqt|Wvt (64K elems each) then Kb|Qb|Vt (1M each)
  __bf16* ws  = (__bf16*)d_ws;
  __bf16* Wkt = ws;
  __bf16* Wqt = Wkt + (size_t)HDIM * CDIM;
  __bf16* Wvt = Wqt + (size_t)HDIM * CDIM;
  __bf16* Kb  = Wvt + (size_t)HDIM * CDIM;
  __bf16* Qb  = Kb + (size_t)BATCH * T_SEQ * HDIM;
  __bf16* Vt  = Qb + (size_t)BATCH * T_SEQ * HDIM;

  hipLaunchKernelGGL(wtrans, dim3(768), dim3(256), 0, stream,
                     Wk, Wq, Wv, Wkt, Wqt, Wvt);
  hipLaunchKernelGGL(qkv_proj, dim3(256), dim3(128), 0, stream,
                     x, Wkt, Wqt, Wvt, Kb, Qb, Vt);
  hipLaunchKernelGGL(attn, dim3(256), dim3(128), 0, stream,
                     Qb, Kb, Vt, out);
}
